// MultiHeadAttention_56083682951759
// MI455X (gfx1250) — matrix operand order repaired and checked by reading, not compiled, not run on hardware
//
#include <hip/hip_runtime.h>

// ---------------------------------------------------------------------------
// MI455X (gfx1250) multi-head attention: bf16 WMMA (f32 accumulate) throughout.
//   B=2, S=2048, HIDDEN=1024, HEADS=16, HEAD_DIM=64, M = B*S = 4096.
// ---------------------------------------------------------------------------

typedef __attribute__((ext_vector_type(16))) __bf16 v16bf;
typedef __attribute__((ext_vector_type(8)))  float  v8f;

union FragBF {
  uint4 q[2];
  v16bf v;
};

__device__ __forceinline__ unsigned short f2bf(float f) {
  union { float f; unsigned u; } x;
  x.f = f;
  unsigned u = x.u + 0x7FFFu + ((x.u >> 16) & 1u);  // round-to-nearest-even
  return (unsigned short)(u >> 16);
}

#if __has_builtin(__builtin_amdgcn_cvt_pk_bf16_f32)
typedef __attribute__((ext_vector_type(2))) __bf16 v2bf;
__device__ __forceinline__ unsigned pack2bf(float x, float y) {
  v2bf r = __builtin_amdgcn_cvt_pk_bf16_f32(x, y);
  return __builtin_bit_cast(unsigned, r);
}
#else
__device__ __forceinline__ unsigned pack2bf(float x, float y) {
  return (unsigned)f2bf(x) | ((unsigned)f2bf(y) << 16);
}
#endif

// Build a v16bf fragment from two contiguous 16-byte chunks (8 bf16 each).
__device__ __forceinline__ v16bf frag_from(const unsigned short* p0,
                                           const unsigned short* p1) {
  FragBF f;
  f.q[0] = *(const uint4*)p0;
  f.q[1] = *(const uint4*)p1;
  return f.v;
}

__device__ __forceinline__ v8f wmma_bf16(v16bf a, v16bf b, v8f c) {
  return __builtin_amdgcn_wmma_f32_16x16x32_bf16(
      false, a, false, b, (short)0, c, false, false);
}

// --- 16-lane butterfly reductions via DPP16 (VALU, no LDS traffic) ---------
template <int CTRL>
__device__ __forceinline__ float dppf(float x) {
  return __builtin_bit_cast(
      float, __builtin_amdgcn_mov_dpp(__builtin_bit_cast(int, x), CTRL, 0xF,
                                      0xF, true));
}
__device__ __forceinline__ float redmax16(float v) {
  v = fmaxf(v, dppf<0xB1>(v));   // quad_perm [1,0,3,2]  (xor 1)
  v = fmaxf(v, dppf<0x4E>(v));   // quad_perm [2,3,0,1]  (xor 2)
  v = fmaxf(v, dppf<0x141>(v));  // row_half_mirror      (crosses 4-groups)
  v = fmaxf(v, dppf<0x140>(v));  // row_mirror           (crosses 8-groups)
  return v;
}
__device__ __forceinline__ float redsum16(float v) {
  v += dppf<0xB1>(v);
  v += dppf<0x4E>(v);
  v += dppf<0x141>(v);
  v += dppf<0x140>(v);
  return v;
}

// ---------------------------------------------------------------------------
// GEMM: C[M,N] = (A[M,K] @ W[N,K]^T + bias) * scale   (M=4096, N=K=1024)
// MODE 0: A f32 (activations), output bf16 head-split [B,H,S,D], scaled
// MODE 1: A bf16 (attention context), output f32 flat [M,N]
// Block: 256 threads = 8 waves; block tile 128x128; wave tile 32x64 (2x4
// WMMA). Double-buffered LDS with register prefetch, one barrier per K-step.
// ---------------------------------------------------------------------------
template <int MODE>
__global__ __launch_bounds__(256) void gemm_kernel(
    const void* __restrict__ Ap, const float* __restrict__ W,
    const float* __restrict__ bias, void* __restrict__ Outp, float scale) {
  constexpr int K = 1024;
  __shared__ __align__(16) unsigned short As[2][128 * 32];
  __shared__ __align__(16) unsigned short Ws[2][128 * 32];

  const int t     = threadIdx.x;
  const int lane  = t & 31;
  const int wid   = t >> 5;
  const int waveM = wid >> 1;  // 0..3  (32-row slices)
  const int waveN = wid & 1;   // 0..1  (64-col slices)
  const int m0 = blockIdx.x * 128;
  const int n0 = blockIdx.y * 128;
  const int l16 = lane & 15;
  const int hi  = lane >> 4;
  const int a8  = hi * 8;    // A-fragment K sub-offset
  const int b16 = hi * 16;   // unused; B fragments use a8 too

  const int ldrow = t >> 3;            // 0..31 within a 4-iteration sweep
  const int ldc4  = (t & 7) * 4;       // 0,4,..,28

  v8f acc[2][4] = {};

  float biasv[4];
#pragma unroll
  for (int j = 0; j < 4; ++j)
    biasv[j] = bias[n0 + waveN * 64 + j * 16 + l16];

  float4 ra[4], rw[4];
  uint2  ra16[4];

  auto ldg = [&](int kb) {
    if (MODE == 0) {
      const float* A = (const float*)Ap;
#pragma unroll
      for (int i = 0; i < 4; ++i)
        ra[i] = *(const float4*)&A[(size_t)(m0 + ldrow + 32 * i) * K + kb + ldc4];
    } else {
      const unsigned short* A = (const unsigned short*)Ap;
#pragma unroll
      for (int i = 0; i < 4; ++i)
        ra16[i] = *(const uint2*)&A[(size_t)(m0 + ldrow + 32 * i) * K + kb + ldc4];
    }
#pragma unroll
    for (int i = 0; i < 4; ++i)
      rw[i] = *(const float4*)&W[(size_t)(n0 + ldrow + 32 * i) * K + kb + ldc4];
  };

  auto sts = [&](int buf) {
#pragma unroll
    for (int i = 0; i < 4; ++i) {
      if (MODE == 0) {
        uint2 u;
        u.x = pack2bf(ra[i].x, ra[i].y);
        u.y = pack2bf(ra[i].z, ra[i].w);
        *(uint2*)&As[buf][(ldrow + 32 * i) * 32 + ldc4] = u;
      } else {
        *(uint2*)&As[buf][(ldrow + 32 * i) * 32 + ldc4] = ra16[i];
      }
      uint2 w;
      w.x = pack2bf(rw[i].x, rw[i].y);
      w.y = pack2bf(rw[i].z, rw[i].w);
      *(uint2*)&Ws[buf][(ldrow + 32 * i) * 32 + ldc4] = w;
    }
  };

  ldg(0);
  sts(0);
  __syncthreads();

  int buf = 0;
  for (int kb = 0; kb < K; kb += 32) {
    if (kb + 32 < K) ldg(kb + 32);
    if (kb + 64 < K) {  // L2 prefetch two tiles ahead (global_prefetch_b8)
      if (MODE == 0)
        __builtin_prefetch(
            &((const float*)Ap)[(size_t)(m0 + ldrow) * K + kb + 64 + ldc4], 0, 0);
      __builtin_prefetch(&W[(size_t)(n0 + ldrow) * K + kb + 64 + ldc4], 0, 0);
    }

    // ---- fragments + WMMA from LDS buffer `buf` ----
    v16bf afr[2], bfr[4];
#pragma unroll
    for (int i = 0; i < 2; ++i) {
      int row = waveM * 32 + i * 16 + l16;
      afr[i] = frag_from(&As[buf][row * 32 + a8], &As[buf][row * 32 + 16 + a8]);
    }
#pragma unroll
    for (int j = 0; j < 4; ++j) {
      int n = waveN * 64 + j * 16 + l16;
      bfr[j] = frag_from(&Ws[buf][n * 32 + a8], &Ws[buf][n * 32 + 16 + a8]);
    }
#pragma unroll
    for (int i = 0; i < 2; ++i)
#pragma unroll
      for (int j = 0; j < 4; ++j)
        acc[i][j] = wmma_bf16(afr[i], bfr[j], acc[i][j]);

    if (kb + 32 < K) sts(buf ^ 1);
    __syncthreads();
    buf ^= 1;
  }

  // ---- epilogue ----
#pragma unroll
  for (int i = 0; i < 2; ++i) {
#pragma unroll
    for (int j = 0; j < 4; ++j) {
#pragma unroll
      for (int r = 0; r < 8; ++r) {
        int m = m0 + waveM * 32 + i * 16 + r + hi * 8;
        int n = n0 + waveN * 64 + j * 16 + l16;
        if (MODE == 0) {
          float v = (acc[i][j][r] + biasv[j]) * scale;
          unsigned short* O = (unsigned short*)Outp;
          int b = m >> 11, s = m & 2047;
          int h = n >> 6, d = n & 63;
          O[(((size_t)(b * 16 + h)) * 2048 + s) * 64 + d] = f2bf(v);
        } else {
          float* O = (float*)Outp;
          O[(size_t)m * 1024 + n] = acc[i][j][r] + biasv[j];
        }
      }
    }
  }
}

// ---------------------------------------------------------------------------
// Flash attention over bf16 Q/K/V in [B,H,S,D] (S=2048, D=64). Q comes
// pre-scaled by 1/sqrt(D). Block: 128 threads = 4 waves; 64 query rows per
// block (16 per wave); key blocks of 32, double-buffered K/V staging.
// Output: bf16 context in [B, S, HIDDEN] layout.
// ---------------------------------------------------------------------------
__global__ __launch_bounds__(128) void flash_kernel(
    const unsigned short* __restrict__ Q, const unsigned short* __restrict__ Kg,
    const unsigned short* __restrict__ Vg, unsigned short* __restrict__ Ctx) {
  constexpr int S = 2048, D = 64;
  __shared__ __align__(16) unsigned short Ks[2][32 * 64];    // [key][d]
  __shared__ __align__(16) unsigned short Vt[2][64 * 32];    // [d][key]
  __shared__ __align__(16) unsigned short Ps[4][16 * 32];    // per-wave P

  const int t    = threadIdx.x;
  const int lane = t & 31;
  const int w    = t >> 5;
  const int l16  = lane & 15;
  const int hi   = lane >> 4;
  const int a8   = hi * 8;
  const int b16  = hi * 16;

  const int qblk = blockIdx.x & 31;   // 32 query blocks per (b,h)
  const int bh   = blockIdx.x >> 5;   // b*16 + h
  const int b    = bh >> 4;
  const int h    = bh & 15;
  const size_t base = (size_t)bh * S * D;

  const int ldrow = t >> 4;           // 0..7  (8 key-rows per sweep of 128 thr)
  const int ldc4  = (t & 15) * 4;     // 0,4,..,60

  // persistent Q fragments for this wave's 16 rows (A-layout, K = d)
  const int qrow = qblk * 64 + w * 16 + l16;
  const unsigned short* qp = Q + base + (size_t)qrow * D;
  const v16bf qf0 = frag_from(qp + a8, qp + 16 + a8);        // d 0..31
  const v16bf qf1 = frag_from(qp + 32 + a8, qp + 48 + a8);   // d 32..63

  v8f ov[4] = {};
  float mrow[8], lrow[8];
#pragma unroll
  for (int r = 0; r < 8; ++r) { mrow[r] = -1e30f; lrow[r] = 0.0f; }

  uint2 rk[4], rv[4];
  auto ldg = [&](int kb) {
#pragma unroll
    for (int i = 0; i < 4; ++i) {
      size_t g = base + (size_t)(kb + ldrow + 8 * i) * D + ldc4;
      rk[i] = *(const uint2*)&Kg[g];
      rv[i] = *(const uint2*)&Vg[g];
    }
  };
  auto sts = [&](int buf) {
#pragma unroll
    for (int i = 0; i < 4; ++i) {
      int row = ldrow + 8 * i;
      *(uint2*)&Ks[buf][row * 64 + ldc4] = rk[i];
      union { uint2 u; unsigned short s[4]; } vu;
      vu.u = rv[i];
#pragma unroll
      for (int jj = 0; jj < 4; ++jj)
        Vt[buf][(ldc4 + jj) * 32 + row] = vu.s[jj];  // transpose on store
    }
  };

  ldg(0);
  sts(0);
  __syncthreads();

  int buf = 0;
  for (int kb = 0; kb < S; kb += 32) {
    if (kb + 32 < S) ldg(kb + 32);

    // ---- scores (Q pre-scaled): two 16x16 C tiles over D=64 ----
    v8f c0 = {}, c1 = {};
    {
      int n = l16;  // keys kb..kb+15
      v16bf bk0 = frag_from(&Ks[buf][n * 64 + a8], &Ks[buf][n * 64 + 16 + a8]);
      v16bf bk1 = frag_from(&Ks[buf][n * 64 + 32 + a8],
                            &Ks[buf][n * 64 + 48 + a8]);
      c0 = wmma_bf16(qf0, bk0, c0);
      c0 = wmma_bf16(qf1, bk1, c0);
      n = 16 + l16;  // keys kb+16..kb+31
      v16bf bk2 = frag_from(&Ks[buf][n * 64 + a8], &Ks[buf][n * 64 + 16 + a8]);
      v16bf bk3 = frag_from(&Ks[buf][n * 64 + 32 + a8],
                            &Ks[buf][n * 64 + 48 + a8]);
      c1 = wmma_bf16(qf0, bk2, c1);
      c1 = wmma_bf16(qf1, bk3, c1);
    }

    // ---- online softmax update (DPP reductions across the 16 N-lanes) ----
#pragma unroll
    for (int r = 0; r < 8; ++r) {
      float s0 = c0[r];
      float s1 = c1[r];
      float cm = redmax16(fmaxf(s0, s1));
      float mnew  = fmaxf(mrow[r], cm);
      float alpha = __expf(mrow[r] - mnew);
      float p0 = __expf(s0 - mnew);
      float p1 = __expf(s1 - mnew);
      float rs = redsum16(p0 + p1);
      lrow[r] = lrow[r] * alpha + rs;
      mrow[r] = mnew;
      ov[0][r] *= alpha; ov[1][r] *= alpha;
      ov[2][r] *= alpha; ov[3][r] *= alpha;
      // C-layout -> LDS so we can re-read in A-layout
      int row = r + hi * 8;
      Ps[w][row * 32 + l16]      = f2bf(p0);
      Ps[w][row * 32 + 16 + l16] = f2bf(p1);
    }

    // ---- O += P @ V (K = 32 keys) ----
    v16bf pf = frag_from(&Ps[w][l16 * 32 + a8], &Ps[w][l16 * 32 + 16 + a8]);
#pragma unroll
    for (int tt = 0; tt < 4; ++tt) {
      int dcol = tt * 16 + l16;
      v16bf vf = frag_from(&Vt[buf][dcol * 32 + a8],
                           &Vt[buf][dcol * 32 + 16 + a8]);
      ov[tt] = wmma_bf16(pf, vf, ov[tt]);
    }

    if (kb + 32 < S) sts(buf ^ 1);
    __syncthreads();
    buf ^= 1;
  }

  // ---- finalize: divide by row sums, write bf16 context [B,S,HIDDEN] ----
#pragma unroll
  for (int r = 0; r < 8; ++r) {
    float inv = 1.0f / lrow[r];
    int s = qblk * 64 + w * 16 + r + hi * 8;
    size_t rowbase = ((size_t)b * 2048 + s) * 1024 + (size_t)h * 64;
#pragma unroll
    for (int tt = 0; tt < 4; ++tt)
      Ctx[rowbase + tt * 16 + l16] = f2bf(ov[tt][r] * inv);
  }
}

// ---------------------------------------------------------------------------
// launch
// ---------------------------------------------------------------------------
extern "C" void kernel_launch(void* const* d_in, const int* in_sizes, int n_in,
                              void* d_out, int out_size, void* d_ws,
                              size_t ws_size, hipStream_t stream) {
  (void)in_sizes; (void)n_in; (void)out_size; (void)ws_size;
  const float* query = (const float*)d_in[0];
  const float* key   = (const float*)d_in[1];
  const float* value = (const float*)d_in[2];
  const float* Wq = (const float*)d_in[3];
  const float* bq = (const float*)d_in[4];
  const float* Wk = (const float*)d_in[5];
  const float* bk = (const float*)d_in[6];
  const float* Wv = (const float*)d_in[7];
  const float* bv = (const float*)d_in[8];
  const float* Wo = (const float*)d_in[9];
  const float* bo = (const float*)d_in[10];

  const size_t MH = (size_t)4096 * 1024;  // elements per bf16 intermediate
  unsigned short* Qb = (unsigned short*)d_ws;
  unsigned short* Kb = Qb + MH;
  unsigned short* Vb = Kb + MH;
  unsigned short* Cb = Vb + MH;           // 32 MB of workspace total

  dim3 gg(4096 / 128, 1024 / 128);
  dim3 gb(256);
  // Q is pre-scaled by 1/sqrt(HEAD_DIM) so flash skips the score scaling.
  gemm_kernel<0><<<gg, gb, 0, stream>>>((const void*)query, Wq, bq, (void*)Qb,
                                        0.125f);
  gemm_kernel<0><<<gg, gb, 0, stream>>>((const void*)key, Wk, bk, (void*)Kb,
                                        1.0f);
  gemm_kernel<0><<<gg, gb, 0, stream>>>((const void*)value, Wv, bv, (void*)Vb,
                                        1.0f);

  flash_kernel<<<dim3(2 * 16 * (2048 / 64)), dim3(128), 0, stream>>>(Qb, Kb, Vb,
                                                                     Cb);

  gemm_kernel<1><<<gg, gb, 0, stream>>>((const void*)Cb, Wo, bo, d_out, 1.0f);
}
